// Decoder_85066122265362
// MI455X (gfx1250) — matrix operand order repaired and checked by reading, not compiled, not run on hardware
//
#include <hip/hip_runtime.h>
#include <hip/hip_bf16.h>

typedef __attribute__((ext_vector_type(16))) _Float16 v16h;
typedef __attribute__((ext_vector_type(8)))  _Float16 v8h;
typedef __attribute__((ext_vector_type(8)))  float    v8f;

#define NP   64     // pedestrians per scene
#define HDIM 32
#define EMBD 16
#define IND  48
#define HID  128
#define OUTD 8

// ---- WMMA fragment helpers (wave32, 16x16x32 f16 layouts per CDNA5 ISA 7.12.2) ----

// A matrix 16x32 f16 tile from row-major LDS (leading dim ld halves).
// Lane layout: row = row0 + (lane&15); K = (lane>>4)*8 + {0..7} and 16 + same.
__device__ __forceinline__ v16h lds_load_afrag(const _Float16* p, int ld, int row0, int k0) {
  const int lane = threadIdx.x & 31;
  const _Float16* rp = p + (row0 + (lane & 15)) * ld + k0 + (lane >> 4) * 8;
  v8h a0 = *(const v8h*)rp;         // K offsets 0..7   -> elements 0..7
  v8h a1 = *(const v8h*)(rp + 16);  // K offsets 16..23 -> elements 8..15
  return __builtin_shufflevector(a0, a1, 0, 1, 2, 3, 4, 5, 6, 7,
                                 8, 9, 10, 11, 12, 13, 14, 15);
}

// B matrix 32x16 f16 tile from column-major LDS ([n][k], leading dim ldk halves).
// Lane layout: col = col0 + (lane&15); K = k0 + (lane>>4)*8 + {0..7} and 16 + same.
__device__ __forceinline__ v16h lds_load_bfragT(const _Float16* p, int ldk, int k0, int col0) {
  const int lane = threadIdx.x & 31;
  const _Float16* cp = p + (col0 + (lane & 15)) * ldk + k0 + (lane >> 4) * 8;
  v8h b0 = *(const v8h*)cp;
  v8h b1 = *(const v8h*)(cp + 16);
  return __builtin_shufflevector(b0, b1, 0, 1, 2, 3, 4, 5, 6, 7,
                                 8, 9, 10, 11, 12, 13, 14, 15);
}

// D f32 16x16 accumulator -> row-major LDS f16 (per-lane scattered b16 stores;
// inherent to D col-per-lane layout feeding a row-major A operand next)
__device__ __forceinline__ void lds_store_dfrag(const v8f& acc, _Float16* p, int ld,
                                                int row0, int col0, bool relu) {
  const int lane = threadIdx.x & 31;
  const int col = col0 + (lane & 15);
  const int hi = lane >> 4;
#pragma unroll
  for (int r = 0; r < 8; ++r) {
    float x = acc[r];
    if (relu) x = x > 0.f ? x : 0.f;
    p[(row0 + r + hi * 8) * ld + col] = (_Float16)x;
  }
}

// D f32 16x16 accumulator -> column-major LDS f16 (one packed b128 store per lane)
__device__ __forceinline__ void lds_store_dfragT(const v8f& acc, _Float16* p, int ldk,
                                                 int row0, int col0, bool relu) {
  const int lane = threadIdx.x & 31;
  const int col = col0 + (lane & 15);
  const int hi = lane >> 4;
  v8h h;
#pragma unroll
  for (int r = 0; r < 8; ++r) {
    float x = acc[r];
    if (relu) x = x > 0.f ? x : 0.f;
    h[r] = (_Float16)x;
  }
  *(v8h*)(p + col * ldk + row0 + hi * 8) = h;
}

__global__ __launch_bounds__(256) void sgcn_fused(
    const float* __restrict__ h_states, const float* __restrict__ end_pos,
    const int* __restrict__ end_group, const float* __restrict__ W_spatial,
    const float* __restrict__ b_spatial, const float* __restrict__ W_intra1,
    const float* __restrict__ W_intra2, const float* __restrict__ W_inter1,
    const float* __restrict__ W_inter2, const float* __restrict__ W_out,
    const float* __restrict__ b_out, float* __restrict__ out)
{
  // 62.25 KB carved from one block; aliases:
  //   sU  : W1^T (H1 phase)  <->  Y2 (Y2/H2 phases)
  //   sY  : Y (Y/H1 phases)  <->  W2^T (Y2/H2 phases)
  __shared__ alignas(16) unsigned char smem[63744];
  _Float16* sA    = (_Float16*)(smem);           //  8 KB row-major [64][64]
  _Float16* sXt   = (_Float16*)(smem + 8192);    //  8 KB col-major [feat 64][ped 64]
  _Float16* sY    = (_Float16*)(smem + 16384);   //  8 KB row-major [64][64] / W2^T [16][128]
  _Float16* sH1t  = (_Float16*)(smem + 24576);   // 16 KB col-major [feat 128][ped 64]
  _Float16* sU    = (_Float16*)(smem + 40960);   // 16 KB W1^T col-major / Y2 row-major
  float*    sFeat = (float*)(smem + 57344);      //  4 KB [64][16]
  float*    sOut  = (float*)(smem + 61440);      //  2 KB [64][8]
  int*      sg    = (int*)(smem + 63488);        // 256 B

  const int tid  = threadIdx.x;
  const int wave = tid >> 5;
  const int s = blockIdx.x >> 6;
  const int i = blockIdx.x & 63;
  const int base = s * NP;

  if (tid < NP) sg[tid] = end_group[base + tid];
  __syncthreads();

  // ---- build X_i^T (col-major): sXt[c*64 + k] = [emb | hidden | 0-pad](ped k, feat c)
  const float pix = end_pos[2 * (base + i)];
  const float piy = end_pos[2 * (base + i) + 1];
  for (int idx = tid; idx < NP * 64; idx += 256) {
    const int k = idx >> 6, c = idx & 63;
    float v = 0.f;
    if (c < EMBD) {
      const float rx = end_pos[2 * (base + k)]     - pix;
      const float ry = end_pos[2 * (base + k) + 1] - piy;
      v = rx * W_spatial[c] + ry * W_spatial[EMBD + c] + b_spatial[c];
    } else if (c < IND) {
      v = h_states[(base + k) * HDIM + (c - EMBD)];
    }
    sXt[c * 64 + k] = (_Float16)v;
  }

  for (int m = 0; m < 2; ++m) {
    // ---- normalized adjacency for mask m:
    // A[j,k] = adj_all[i,j,k]*mask[j,k] / ((j==k) ? rowsum(mask[j,:]) : 1+mask[j,k])
    if (tid < NP) {
      const int j = tid;
      const int gj = sg[j];
      float rs = 0.f;
      for (int k = 0; k < NP; ++k) {
        const bool same = (gj == sg[k]) && (gj != 0);
        const bool ms = same || (j == k);
        const bool mm = m ? ((!ms) || (j == k)) : ms;
        rs += mm ? 1.f : 0.f;
      }
      for (int k = 0; k < NP; ++k) {
        const bool same = (gj == sg[k]) && (gj != 0);
        const bool ms = same || (j == k);
        const bool mm = m ? ((!ms) || (j == k)) : ms;
        const bool adj = (j == k) || (k == i) || (j == i);
        const float mv = mm ? 1.f : 0.f;
        const float d  = (j == k) ? rs : (1.f + mv);
        sA[j * NP + k] = (_Float16)(adj ? (mv / d) : 0.f);
      }
    }
    // ---- stage W1^T into sU (coalesced global read, transposed LDS write)
    {
      const float* W1 = m ? W_inter1 : W_intra1;
      for (int idx = tid; idx < 64 * HID; idx += 256) {
        const int k = idx >> 7, c = idx & 127;   // k: input feat 0..63, c: out feat
        const float v = (k < IND) ? W1[k * HID + c] : 0.f;
        sU[c * 64 + k] = (_Float16)v;
      }
    }
    __syncthreads();

    // ---- Y = A @ X : -> sY row-major (ld 64), 16 tiles, 2/wave ----
    for (int t = wave * 2; t < wave * 2 + 2; ++t) {
      const int r0 = (t >> 2) << 4, c0 = (t & 3) << 4;
      v8f acc = {};
#pragma unroll
      for (int kk = 0; kk < 2; ++kk) {
        v16h a = lds_load_afrag(sA, NP, r0, kk * 32);
        v16h b = lds_load_bfragT(sXt, 64, kk * 32, c0);
        acc = __builtin_amdgcn_wmma_f32_16x16x32_f16(false, a, false, b,
                                                     (short)0, acc, false, false);
      }
      lds_store_dfrag(acc, sY, 64, r0, c0, false);
    }
    __syncthreads();

    // ---- H1 = relu(Y @ W1) : -> sH1t col-major, 32 tiles, 4/wave ----
    for (int t = wave * 4; t < wave * 4 + 4; ++t) {
      const int r0 = (t >> 3) << 4, c0 = (t & 7) << 4;
      v8f acc = {};
#pragma unroll
      for (int kk = 0; kk < 2; ++kk) {
        v16h a = lds_load_afrag(sY, 64, r0, kk * 32);
        v16h b = lds_load_bfragT(sU, 64, kk * 32, c0);   // W1^T
        acc = __builtin_amdgcn_wmma_f32_16x16x32_f16(false, a, false, b,
                                                     (short)0, acc, false, false);
      }
      lds_store_dfragT(acc, sH1t, 64, r0, c0, true);
    }
    __syncthreads();

    // ---- stage W2^T into sY region (Y is dead; 4 KB coalesced read, pad to 16 cols)
    {
      const float* W2 = m ? W_inter2 : W_intra2;
      for (int idx = tid; idx < 16 * HID; idx += 256) {
        const int k = idx >> 4, c = idx & 15;    // k: hidden feat 0..127, c: out feat
        const float v = (c < OUTD) ? W2[k * OUTD + c] : 0.f;
        sY[c * HID + k] = (_Float16)v;           // col-major [16][128]
      }
    }
    // ---- Y2 = A @ H1 : -> sU row-major (ld 128), 32 tiles, 4/wave ----
    for (int t = wave * 4; t < wave * 4 + 4; ++t) {
      const int r0 = (t >> 3) << 4, c0 = (t & 7) << 4;
      v8f acc = {};
#pragma unroll
      for (int kk = 0; kk < 2; ++kk) {
        v16h a = lds_load_afrag(sA, NP, r0, kk * 32);
        v16h b = lds_load_bfragT(sH1t, 64, kk * 32, c0);
        acc = __builtin_amdgcn_wmma_f32_16x16x32_f16(false, a, false, b,
                                                     (short)0, acc, false, false);
      }
      lds_store_dfrag(acc, sU, HID, r0, c0, false);      // Y2 (W1t now dead)
    }
    __syncthreads();

    // ---- H2 = relu(Y2 @ W2) : [64x128]x[128x8 pad 16] -> sFeat[:, m*8 + c] ----
    if (wave < 4) {
      const int r0 = wave << 4;
      v8f acc = {};
#pragma unroll
      for (int kk = 0; kk < 4; ++kk) {
        v16h a = lds_load_afrag(sU, HID, r0, kk * 32);
        v16h b = lds_load_bfragT(sY, HID, kk * 32, 0);   // W2^T
        acc = __builtin_amdgcn_wmma_f32_16x16x32_f16(false, a, false, b,
                                                     (short)0, acc, false, false);
      }
      const int lane = tid & 31;
      const int col = lane & 15, hi = lane >> 4;
      if (col < OUTD) {
#pragma unroll
        for (int r = 0; r < 8; ++r) {
          float x = acc[r];
          x = x > 0.f ? x : 0.f;
          sFeat[(r0 + r + hi * 8) * 16 + m * OUTD + col] = x;
        }
      }
    }
    __syncthreads();
  }

  // ---- out[j] = feat[j] @ W_out + b_out ; then max over j ----
  for (int t = tid; t < NP * OUTD; t += 256) {
    const int j = t >> 3, o = t & 7;
    float acc = b_out[o];
#pragma unroll
    for (int c = 0; c < 16; ++c) acc += sFeat[j * 16 + c] * W_out[c * OUTD + o];
    sOut[t] = acc;
  }
  __syncthreads();

  if (tid < OUTD) {
    float mx = -3.402823466e38f;
    for (int j = 0; j < NP; ++j) {
      const float v = sOut[j * OUTD + tid];
      mx = v > mx ? v : mx;
    }
    out[blockIdx.x * OUTD + tid] = mx;
  }
}

extern "C" void kernel_launch(void* const* d_in, const int* in_sizes, int n_in,
                              void* d_out, int out_size, void* d_ws, size_t ws_size,
                              hipStream_t stream) {
  const float* h_states  = (const float*)d_in[0];
  // d_in[1] = seq_start_end (contiguous blocks of 64 by construction; unused)
  const float* end_pos   = (const float*)d_in[2];
  const int*   end_group = (const int*)d_in[3];
  const float* W_spatial = (const float*)d_in[4];
  const float* b_spatial = (const float*)d_in[5];
  const float* W_intra1  = (const float*)d_in[6];
  const float* W_intra2  = (const float*)d_in[7];
  const float* W_inter1  = (const float*)d_in[8];
  const float* W_inter2  = (const float*)d_in[9];
  const float* W_out     = (const float*)d_in[10];
  const float* b_out     = (const float*)d_in[11];
  float* out = (float*)d_out;

  const int B = in_sizes[2] / 2;   // S*N = 8192, one workgroup per (scene, pedestrian)
  sgcn_fused<<<dim3(B), dim3(256), 0, stream>>>(
      h_states, end_pos, end_group, W_spatial, b_spatial,
      W_intra1, W_intra2, W_inter1, W_inter2, W_out, b_out, out);
}
